// myxLSTM_87711822119475
// MI455X (gfx1250) — matrix operand order repaired and checked by reading, not compiled, not run on hardware
//
#include <hip/hip_runtime.h>
#include <hip/hip_bf16.h>
#include <math.h>

// ---------------- problem constants ----------------
constexpr int Bb  = 256;   // batch
constexpr int L   = 4096;
constexpr int CIN = 2;
constexpr int D   = 128;
constexpr int H   = 4;
constexpr int DH  = 32;
constexpr int FF  = 192;
constexpr int NC  = 5;
constexpr int P1  = 2047;  // after conv1(4095)+pool
constexpr int P2  = 1022;
constexpr int L3  = 1020;
constexpr int S   = 510;
constexpr float EPS = 1e-5f;

// ---------------- workspace layout (floats) ----------------
// region A: h1 (B,32,P1)      -> later xn -> later ybuf
// region B: h2 (B,64,P2)      -> later xcv -> later xn2
// region E: e  (B,S,D)        residual stream (persistent)
// region G: gx (S,B,4,H,DH)   -> later up (B*S,384)
constexpr size_t A_OFF = 0;
constexpr size_t A_SZ  = (size_t)Bb * 32 * P1;
constexpr size_t B_OFF = A_OFF + A_SZ;
constexpr size_t B_SZ  = (size_t)Bb * 64 * P2;
constexpr size_t E_OFF = B_OFF + B_SZ;
constexpr size_t E_SZ  = (size_t)Bb * S * D;
constexpr size_t G_OFF = E_OFF + E_SZ;
// total ws needed ~469 MB

// ---------------- WMMA bf16 plumbing (wave32, gfx1250) ----------------
typedef __bf16 bf16;
typedef __attribute__((ext_vector_type(16))) __bf16 v16bf;
typedef __attribute__((ext_vector_type(8)))  float  v8f;
typedef __attribute__((ext_vector_type(4)))  unsigned int v4u;
typedef __attribute__((ext_vector_type(8)))  int v8i;
typedef __attribute__((ext_vector_type(4)))  int v4i;

// A fragment: tile row-major [16][ld] bf16 in LDS.
// lanes 0-15: M=lane, elems 0..7 = K kb+0..7, elems 8..15 = K kb+16..23
// lanes 16-31: M=lane-16, K halves +8 / +24   (ISA 16-bit A 16x32 layout)
__device__ inline v16bf frag_a(const bf16* tile, int ld, int kb, int lane) {
  int r = lane & 15, hi = (lane >> 4) & 1;
  const bf16* p = tile + r * ld + kb + hi * 8;
  v16bf a;
#pragma unroll
  for (int j = 0; j < 8; ++j) { a[j] = p[j]; a[8 + j] = p[16 + j]; }
  return a;
}
// Same, but from an f32 LDS tile (TDM-staged), converting to bf16 at load.
__device__ inline v16bf frag_a_f32(const float* tile, int ld, int kb, int lane) {
  int r = lane & 15, hi = (lane >> 4) & 1;
  const float* p = tile + r * ld + kb + hi * 8;
  v16bf a;
#pragma unroll
  for (int j = 0; j < 8; ++j) { a[j] = (bf16)p[j]; a[8 + j] = (bf16)p[16 + j]; }
  return a;
}
// B fragment from transposed weights Wt[n][k] in LDS:
// lane n<16 holds K kb+0..7, kb+16..23 of column n; lanes 16-31 hold K kb+8..15, kb+24..31
__device__ inline v16bf frag_b(const bf16* wt, int ld, int kb, int lane) {
  int n = lane & 15, hi = (lane >> 4) & 1;
  const bf16* p = wt + n * ld + kb + hi * 8;
  v16bf b;
#pragma unroll
  for (int j = 0; j < 8; ++j) { b[j] = p[j]; b[8 + j] = p[16 + j]; }
  return b;
}
__device__ inline v8f wmma_bf16(v16bf a, v16bf b, v8f c) {
  return __builtin_amdgcn_wmma_f32_16x16x32_bf16(false, a, false, b, (short)0, c,
                                                 false, false);
}

__device__ inline float wave_sum32(float v) {
#pragma unroll
  for (int o = 16; o > 0; o >>= 1) v += __shfl_xor(v, o, 32);
  return v;
}
__device__ inline float leakyf(float x) { return x >= 0.f ? x : 0.01f * x; }

// ---------------- TDM: async 2D f32 tile load (global -> LDS), D# built per ISA 08 §8 ----
// tile_rows x tile_k f32 elements, row stride row_stride elements, compacted into LDS.
__device__ inline void tdm_load_tile_f32(unsigned int lds_off, const float* gptr,
                                         int tile_k, int tile_rows, int row_stride) {
  unsigned long long ga = (unsigned long long)(uintptr_t)gptr;
  unsigned int d0 = (unsigned int)row_stride;   // tensor_dim0 >= tile_k: no x-clip
  unsigned int d1 = 1u << 20;                   // tensor_dim1: large, no y-clip
  v4u g0;
  g0[0] = 1u;                                            // count=1, user D#
  g0[1] = lds_off;                                       // lds_addr (bytes)
  g0[2] = (unsigned int)(ga & 0xffffffffu);              // global_addr[31:0]
  g0[3] = (unsigned int)((ga >> 32) & 0x01ffffffu)       // global_addr[56:32]
          | (2u << 30);                                  // type=2 ("image")
  v8i g1;
  g1[0] = (int)(2u << 16);                               // data_size=4B, mask/flags=0
  g1[1] = (int)((d0 & 0xffffu) << 16);                   // tensor_dim0[15:0]
  g1[2] = (int)((d0 >> 16) | ((d1 & 0xffffu) << 16));    // dim0[31:16] | dim1[15:0]
  g1[3] = (int)((d1 >> 16) | (((unsigned)tile_k) << 16));// dim1[31:16] | tile_dim0
  g1[4] = (int)((unsigned)tile_rows & 0xffffu);          // tile_dim1 (tile_dim2=0)
  g1[5] = (int)row_stride;                               // tensor_dim0_stride[31:0]
  g1[6] = 0;                                             // stride0[47:32] | stride1 lo
  g1[7] = 0;
  v4i gz = {0, 0, 0, 0};
#if __clang_major__ >= 23
  v8i gz8 = {0, 0, 0, 0, 0, 0, 0, 0};
  __builtin_amdgcn_tensor_load_to_lds(g0, g1, gz, gz, gz8, 0);
#else
  __builtin_amdgcn_tensor_load_to_lds(g0, g1, gz, gz, 0);
#endif
}

// ---------------- K1: conv1(w1a k4 pad1 || w1b k2) + BN + leaky + pool ----------------
__global__ void k_conv1(const float* __restrict__ x, const float* __restrict__ w1a,
                        const float* __restrict__ w1b, const float* __restrict__ g1,
                        const float* __restrict__ b1, const float* __restrict__ m1,
                        const float* __restrict__ v1, float* __restrict__ h1) {
  __shared__ float swa[128], swb[64], ssc[32], ssh[32];
  int tid = threadIdx.x;
  if (tid < 128) swa[tid] = w1a[tid];
  if (tid < 64)  swb[tid] = w1b[tid];
  if (tid < 32) {
    float sc = g1[tid] * rsqrtf(v1[tid] + EPS);
    ssc[tid] = sc; ssh[tid] = b1[tid] - m1[tid] * sc;
  }
  __syncthreads();
  int tp = blockIdx.x * 256 + tid;
  int b  = blockIdx.y;
  if (tp >= P1) return;
  float xv[5][2];
  const float* xb = x + (size_t)b * L * CIN;
#pragma unroll
  for (int i = 0; i < 5; ++i) {
    int l = 2 * tp - 1 + i;
    bool ok = (l >= 0) && (l < L);
    xv[i][0] = ok ? xb[l * 2 + 0] : 0.f;
    xv[i][1] = ok ? xb[l * 2 + 1] : 0.f;
  }
  float* out = h1 + (size_t)b * 32 * P1 + tp;
#pragma unroll
  for (int c = 0; c < 16; ++c) {
    float a0 = 0.f, a1 = 0.f;
#pragma unroll
    for (int ci = 0; ci < 2; ++ci)
#pragma unroll
      for (int k = 0; k < 4; ++k) {
        float w = swa[(c * 2 + ci) * 4 + k];
        a0 += xv[k][ci] * w;
        a1 += xv[k + 1][ci] * w;
      }
    a0 = leakyf(a0 * ssc[c] + ssh[c]);
    a1 = leakyf(a1 * ssc[c] + ssh[c]);
    out[(size_t)c * P1] = fmaxf(a0, a1);
  }
#pragma unroll
  for (int c = 0; c < 16; ++c) {
    float a0 = 0.f, a1 = 0.f;
#pragma unroll
    for (int ci = 0; ci < 2; ++ci)
#pragma unroll
      for (int k = 0; k < 2; ++k) {
        float w = swb[(c * 2 + ci) * 2 + k];
        a0 += xv[k + 1][ci] * w;
        a1 += xv[k + 2][ci] * w;
      }
    int cc = 16 + c;
    a0 = leakyf(a0 * ssc[cc] + ssh[cc]);
    a1 = leakyf(a1 * ssc[cc] + ssh[cc]);
    out[(size_t)cc * P1] = fmaxf(a0, a1);
  }
}

// ---------------- K2: conv2 (64,32,3) + BN + leaky + pool ----------------
__global__ void k_conv2(const float* __restrict__ h1, const float* __restrict__ pw2,
                        const float* __restrict__ g2, const float* __restrict__ b2,
                        const float* __restrict__ m2, const float* __restrict__ v2,
                        float* __restrict__ h2) {
  __shared__ float sw[64 * 32 * 3];
  __shared__ float sx[32][258];
  __shared__ float ssc[64], ssh[64];
  int tid = threadIdx.x;            // 128
  int b = blockIdx.y;
  int tp0 = blockIdx.x * 128;
  for (int i = tid; i < 6144; i += 128) sw[i] = pw2[i];
  if (tid < 64) {
    float sc = g2[tid] * rsqrtf(v2[tid] + EPS);
    ssc[tid] = sc; ssh[tid] = b2[tid] - m2[tid] * sc;
  }
  for (int i = tid; i < 32 * 258; i += 128) {
    int ci = i / 258, p = i % 258;
    int pos = 2 * tp0 + p;
    sx[ci][p] = (pos < P1) ? h1[((size_t)b * 32 + ci) * P1 + pos] : 0.f;
  }
  __syncthreads();
  int tp = tp0 + tid;
  if (tp >= P2) return;
  float* out = h2 + (size_t)b * 64 * P2 + tp;
  for (int c = 0; c < 64; ++c) {
    float a0 = 0.f, a1 = 0.f;
    const float* wc = sw + c * 96;
#pragma unroll
    for (int ci = 0; ci < 32; ++ci) {
      float x0 = sx[ci][2 * tid], x1 = sx[ci][2 * tid + 1];
      float x2 = sx[ci][2 * tid + 2], x3 = sx[ci][2 * tid + 3];
      float w0 = wc[ci * 3], w1 = wc[ci * 3 + 1], w2c = wc[ci * 3 + 2];
      a0 += x0 * w0 + x1 * w1 + x2 * w2c;
      a1 += x1 * w0 + x2 * w1 + x3 * w2c;
    }
    a0 = leakyf(a0 * ssc[c] + ssh[c]);
    a1 = leakyf(a1 * ssc[c] + ssh[c]);
    out[(size_t)c * P2] = fmaxf(a0, a1);
  }
}

// ---------------- K3: conv3 as im2col GEMM via WMMA bf16 + BN + leaky + pool ----------------
__global__ void k_conv3_wmma(const float* __restrict__ h2, const float* __restrict__ pw3,
                             const float* __restrict__ g3, const float* __restrict__ b3,
                             const float* __restrict__ m3, const float* __restrict__ v3,
                             float* __restrict__ e) {
  __shared__ bf16 sW[128 * 192];   // Wt[n][kk]; w3 (c,ci,k) row-major == [n][kk]
  __shared__ bf16 sA[16 * 192];
  int tid = threadIdx.x;            // 256
  int b = blockIdx.y;
  int t0 = blockIdx.x * 16;
  for (int i = tid; i < 128 * 192; i += 256) sW[i] = (bf16)pw3[i];
  for (int i = tid; i < 16 * 192; i += 256) {
    int row = i / 192, kk = i % 192, ci = kk / 3, k = kk % 3;
    int t = t0 + row;
    float val = (t < L3) ? h2[((size_t)b * 64 + ci) * P2 + t + k] : 0.f;
    sA[i] = (bf16)val;
  }
  __syncthreads();
  int wave = tid >> 5, lane = tid & 31;
  int n0 = wave * 16;
  v8f acc = {0.f, 0.f, 0.f, 0.f, 0.f, 0.f, 0.f, 0.f};
#pragma unroll
  for (int kb = 0; kb < 6; ++kb) {
    v16bf a  = frag_a(sA, 192, kb * 32, lane);
    v16bf bb = frag_b(sW + n0 * 192, 192, kb * 32, lane);
    acc = wmma_bf16(a, bb, acc);
  }
  int col = n0 + (lane & 15);
  int hi = lane >> 4;
  float scale = g3[col] * rsqrtf(v3[col] + EPS);
  float shift = b3[col] - m3[col] * scale;
#pragma unroll
  for (int j = 0; j < 4; ++j) {
    int t = t0 + hi * 8 + 2 * j;
    if (t < L3) {
      float a0 = leakyf(acc[2 * j] * scale + shift);
      float a1 = leakyf(acc[2 * j + 1] * scale + shift);
      int sp = t >> 1;
      e[((size_t)b * S + sp) * D + col] = fmaxf(a0, a1);
    }
  }
}

// ---------------- LayerNorm (no bias) over D=128 ----------------
__global__ void k_layernorm(const float* __restrict__ src, float* __restrict__ dst,
                            const float* __restrict__ w) {
  size_t r = blockIdx.x;
  int c = threadIdx.x;               // 128
  __shared__ float red[8];
  float v = src[r * D + c];
  float s = wave_sum32(v);
  if ((c & 31) == 0) red[c >> 5] = s;
  __syncthreads();
  float mu = (red[0] + red[1] + red[2] + red[3]) * (1.f / D);
  float d0 = v - mu;
  float s2 = wave_sum32(d0 * d0);
  if ((c & 31) == 0) red[4 + (c >> 5)] = s2;
  __syncthreads();
  float var = (red[4] + red[5] + red[6] + red[7]) * (1.f / D);
  dst[r * D + c] = d0 * rsqrtf(var + EPS) * w[c];
}

// ---------------- depthwise causal conv (k=3, pad left 2) + SiLU ----------------
__global__ void k_dwconv_silu(const float* __restrict__ xn, const float* __restrict__ cw,
                              const float* __restrict__ cb, float* __restrict__ xcv) {
  size_t idx = (size_t)blockIdx.x * 256 + threadIdx.x;
  if (idx >= (size_t)Bb * S * D) return;
  int c = idx & (D - 1);
  size_t bt = idx >> 7;
  int t = (int)(bt % S);
  size_t brow = bt - t;
  float acc = cb[c];
#pragma unroll
  for (int k = 0; k < 3; ++k) {
    int tt = t - 2 + k;
    if (tt >= 0) acc += xn[(brow + tt) * D + c] * cw[c * 3 + k];
  }
  xcv[idx] = acc / (1.f + __expf(-acc));
}

// ---------------- gate projections (i,f from xcv; z,o from xn) via WMMA, + rbias ----------------
__global__ void k_gates_wmma(const float* __restrict__ xn, const float* __restrict__ xcv,
                             const float* __restrict__ wi, const float* __restrict__ wf,
                             const float* __restrict__ wz, const float* __restrict__ wo,
                             const float* __restrict__ rbias, float* __restrict__ gx) {
  __shared__ bf16 sW[16 * 1024];   // [g*4+h][e][d] transposed
  __shared__ float sB[512];        // rbias (4,H,32)
  __shared__ bf16 sXc[16 * 128], sXn[16 * 128];
  int tid = threadIdx.x;            // 256
  const float* wsrc[4] = {wi, wf, wz, wo};
  for (int i = tid; i < 16 * 1024; i += 256) {
    int gh = i >> 10, g = gh >> 2, h = gh & 3;
    int rem = i & 1023, e_ = rem >> 5, dd = rem & 31;
    sW[i] = (bf16)wsrc[g][h * 1024 + dd * 32 + e_];
  }
  for (int i = tid; i < 512; i += 256) sB[i] = rbias[i];
  int wave = tid >> 5, lane = tid & 31;
  for (int mt = 0; mt < 8; ++mt) {
    size_t row0 = ((size_t)blockIdx.x * 8 + mt) * 16;
    __syncthreads();
    for (int i = tid; i < 16 * 128; i += 256) {
      size_t r = row0 + (i >> 7);
      sXc[i] = (bf16)xcv[r * D + (i & 127)];
      sXn[i] = (bf16)xn[r * D + (i & 127)];
    }
    __syncthreads();
#pragma unroll
    for (int j = 0; j < 4; ++j) {
      int job = wave * 4 + j;                 // 32 jobs = 16 (h,g) x 2 N-tiles
      int nt = job & 1, gh = job >> 1;
      int g = gh >> 2, h = gh & 3;
      const bf16* src = (g < 2) ? sXc : sXn;
      v16bf a  = frag_a(src + h * 32, 128, 0, lane);
      v16bf bb = frag_b(sW + gh * 1024 + nt * 512, 32, 0, lane);
      int col = nt * 16 + (lane & 15);
      float bias = sB[(g * 4 + h) * 32 + col];
      v8f c = {bias, bias, bias, bias, bias, bias, bias, bias};
      c = wmma_bf16(a, bb, c);
      int hi = lane >> 4;
#pragma unroll
      for (int i2 = 0; i2 < 8; ++i2) {
        size_t r = row0 + hi * 8 + i2;
        int b = (int)(r / S), t = (int)(r % S);
        gx[(((size_t)t * Bb + b) * 4 + g) * 128 + h * 32 + col] = c[i2];
      }
    }
  }
}

// ---------------- the sequential xLSTM scan: one wave per (batch,head) ----------------
__global__ void k_scan(const float* __restrict__ gx, const float* __restrict__ rkernel,
                       float* __restrict__ ybuf) {
  __shared__ float srk[H * DH * 4 * DH];   // 64KB: rk[h][d][g][e]
  int tid = threadIdx.x;                    // 256
  for (int i = tid; i < H * DH * 4 * DH; i += 256) srk[i] = rkernel[i];
  __syncthreads();
  int wave = tid >> 5, lane = tid & 31;
  int pair = blockIdx.x * 8 + wave;         // 1024 (b,h) pairs
  int b = pair >> 2, h = pair & 3;
  const float* rkh = srk + h * 4096;
  float y = 0.f, cst = 0.f, nst = 0.f, mst = 0.f;
  for (int t = 0; t < S; ++t) {
    const float* gp = gx + (((size_t)t * Bb + b) * 4) * 128 + h * 32 + lane;
    // hide HBM/L2 latency of the serial chain: prefetch gates 6 steps ahead
    if (t + 6 < S)
      __builtin_prefetch(gx + (((size_t)(t + 6) * Bb + b) * 4) * 128 + h * 32 + lane,
                         0, 1);
    float ri = gp[0], rf = gp[128], rz = gp[256], ro = gp[384];
#pragma unroll 4
    for (int d = 0; d < 32; ++d) {
      float yd = __shfl(y, d, 32);
      const float* rk = rkh + d * 128 + lane;
      ri += yd * rk[0];
      rf += yd * rk[32];
      rz += yd * rk[64];
      ro += yd * rk[96];
    }
    float ls = fminf(rf, 0.f) - log1pf(__expf(-fabsf(rf)));  // log_sigmoid
    float lf = mst + ls;
    float mnew = fmaxf(ri, lf);
    float ig = __expf(ri - mnew);
    float fg = __expf(lf - mnew);
    cst = fg * cst + ig * tanhf(rz);
    nst = fg * nst + ig;
    mst = mnew;
    y = (cst / nst) / (1.f + __expf(-ro));
    ybuf[((size_t)b * S + t) * D + h * 32 + lane] = y;
  }
}

// ---------------- per-head GroupNorm over DH + gn_w scale + residual into e ----------------
__global__ void k_gnorm_residual(const float* __restrict__ ybuf,
                                 const float* __restrict__ gnw, float* __restrict__ e) {
  size_t r = blockIdx.x;
  int c = threadIdx.x;                      // 128: each wave == one head
  float v = ybuf[r * D + c];
  float mu = wave_sum32(v) * (1.f / DH);
  float d0 = v - mu;
  float var = wave_sum32(d0 * d0) * (1.f / DH);
  e[r * D + c] += d0 * rsqrtf(var + EPS) * gnw[c];
}

// ---------------- FFN up GEMM (K=128, N=384) via WMMA ----------------
__global__ void k_ffn_up_wmma(const float* __restrict__ xn2, const float* __restrict__ wup,
                              float* __restrict__ up) {
  __shared__ bf16 sW[128 * 128];   // Wt chunk [n][k]
  __shared__ bf16 sA[16 * 128];
  int tid = threadIdx.x, wave = tid >> 5, lane = tid & 31;
  for (int ch = 0; ch < 3; ++ch) {
    int ncol0 = ch * 128;
    __syncthreads();
    for (int i = tid; i < 128 * 128; i += 256) {
      int n = i >> 7, k = i & 127;
      sW[i] = (bf16)wup[k * 384 + ncol0 + n];
    }
    for (int mt = 0; mt < 8; ++mt) {
      __syncthreads();
      size_t row0 = ((size_t)blockIdx.x * 8 + mt) * 16;
      for (int i = tid; i < 16 * 128; i += 256)
        sA[i] = (bf16)xn2[(row0 + (i >> 7)) * D + (i & 127)];
      __syncthreads();
      int n0 = wave * 16;
      v8f c = {0.f, 0.f, 0.f, 0.f, 0.f, 0.f, 0.f, 0.f};
#pragma unroll
      for (int kb = 0; kb < 4; ++kb) {
        v16bf a  = frag_a(sA, 128, kb * 32, lane);
        v16bf bb = frag_b(sW + n0 * 128, 128, kb * 32, lane);
        c = wmma_bf16(a, bb, c);
      }
      int col = ncol0 + n0 + (lane & 15);
      int hi = lane >> 4;
#pragma unroll
      for (int i2 = 0; i2 < 8; ++i2)
        up[(row0 + hi * 8 + i2) * 384 + col] = c[i2];
    }
  }
}

// ---------------- GEGLU: up[:, :FF] = gelu_exact(up[:, :FF]) * up[:, FF:] ----------------
__global__ void k_geglu(float* __restrict__ up) {
  size_t idx = (size_t)blockIdx.x * 256 + threadIdx.x;
  if (idx >= (size_t)Bb * S * FF) return;
  size_t r = idx / FF;
  int j = (int)(idx % FF);
  float a = up[r * 384 + j];
  float g = up[r * 384 + FF + j];
  float ge = 0.5f * a * (1.f + erff(a * 0.7071067811865475f));
  up[r * 384 + j] = ge * g;
}

// ---------------- FFN down GEMM (K=192, N=128) via WMMA, += residual into e --------------
// A tiles (16 x 192 f32, row stride 384) staged global->LDS by the Tensor Data Mover.
__global__ void k_ffn_down_wmma(const float* __restrict__ gact, const float* __restrict__ wdn,
                                float* __restrict__ e) {
  __shared__ bf16  sW[128 * 192];  // Wt [n][k]  48KB
  __shared__ float sAf[16 * 192];  // f32 A tile 12KB (TDM destination)
  int tid = threadIdx.x, wave = tid >> 5, lane = tid & 31;
  for (int i = tid; i < 128 * 192; i += 256) {
    int n = i / 192, k = i % 192;
    sW[i] = (bf16)wdn[k * 128 + n];
  }
  unsigned int lds_off = (unsigned int)(uintptr_t)(&sAf[0]);  // LDS addr = flat[31:0]
  for (int mt = 0; mt < 8; ++mt) {
    size_t row0 = ((size_t)blockIdx.x * 8 + mt) * 16;
    __syncthreads();                       // prior compute done; sAf reusable
    if (wave == 0) {                       // one TDM issue per workgroup
      tdm_load_tile_f32(lds_off, gact + row0 * 384, 192, 16, 384);
      __builtin_amdgcn_s_wait_tensorcnt(0);
    }
    __syncthreads();                       // tile visible to all waves
    int n0 = wave * 16;
    v8f c = {0.f, 0.f, 0.f, 0.f, 0.f, 0.f, 0.f, 0.f};
#pragma unroll
    for (int kb = 0; kb < 6; ++kb) {
      v16bf a  = frag_a_f32(sAf, 192, kb * 32, lane);
      v16bf bb = frag_b(sW + n0 * 192, 192, kb * 32, lane);
      c = wmma_bf16(a, bb, c);
    }
    int col = n0 + (lane & 15), hi = lane >> 4;
#pragma unroll
    for (int i2 = 0; i2 < 8; ++i2) {
      size_t r = row0 + hi * 8 + i2;
      e[r * D + col] += c[i2];
    }
  }
}

// ---------------- final LN (last timestep only) + FC head ----------------
__global__ void k_head(const float* __restrict__ e, const float* __restrict__ lnpw,
                       const float* __restrict__ fcw, const float* __restrict__ fcb,
                       float* __restrict__ out) {
  int b = blockIdx.x, c = threadIdx.x;      // 128
  __shared__ float red[8];
  __shared__ float vn[128];
  float v = e[((size_t)b * S + (S - 1)) * D + c];
  float s = wave_sum32(v);
  if ((c & 31) == 0) red[c >> 5] = s;
  __syncthreads();
  float mu = (red[0] + red[1] + red[2] + red[3]) * (1.f / D);
  float d0 = v - mu;
  float s2 = wave_sum32(d0 * d0);
  if ((c & 31) == 0) red[4 + (c >> 5)] = s2;
  __syncthreads();
  float var = (red[4] + red[5] + red[6] + red[7]) * (1.f / D);
  vn[c] = d0 * rsqrtf(var + EPS) * lnpw[c];
  __syncthreads();
  if (c < NC) {
    float acc = fcb[c];
#pragma unroll 16
    for (int k = 0; k < D; ++k) acc += vn[k] * fcw[k * NC + c];
    out[b * NC + c] = acc;
  }
}

// ---------------- launcher ----------------
extern "C" void kernel_launch(void* const* d_in, const int* in_sizes, int n_in,
                              void* d_out, int out_size, void* d_ws, size_t ws_size,
                              hipStream_t stream) {
  (void)in_sizes; (void)n_in; (void)out_size; (void)ws_size;
  const float* x      = (const float*)d_in[0];
  const float* w1a    = (const float*)d_in[1];
  const float* w1b    = (const float*)d_in[2];
  const float* bn1_g  = (const float*)d_in[3];
  const float* bn1_b  = (const float*)d_in[4];
  const float* bn1_m  = (const float*)d_in[5];
  const float* bn1_v  = (const float*)d_in[6];
  const float* w2     = (const float*)d_in[7];
  const float* bn2_g  = (const float*)d_in[8];
  const float* bn2_b  = (const float*)d_in[9];
  const float* bn2_m  = (const float*)d_in[10];
  const float* bn2_v  = (const float*)d_in[11];
  const float* w3     = (const float*)d_in[12];
  const float* bn3_g  = (const float*)d_in[13];
  const float* bn3_b  = (const float*)d_in[14];
  const float* bn3_m  = (const float*)d_in[15];
  const float* bn3_v  = (const float*)d_in[16];
  const float* ln1_w  = (const float*)d_in[17];
  const float* conv_w = (const float*)d_in[18];
  const float* conv_b = (const float*)d_in[19];
  const float* wi     = (const float*)d_in[20];
  const float* wf     = (const float*)d_in[21];
  const float* wz     = (const float*)d_in[22];
  const float* wo     = (const float*)d_in[23];
  const float* rkern  = (const float*)d_in[24];
  const float* rbias  = (const float*)d_in[25];
  const float* gn_w   = (const float*)d_in[26];
  const float* ln2_w  = (const float*)d_in[27];
  const float* ffn_up = (const float*)d_in[28];
  const float* ffn_dn = (const float*)d_in[29];
  const float* lnp_w  = (const float*)d_in[30];
  const float* fc_w   = (const float*)d_in[31];
  const float* fc_b   = (const float*)d_in[32];

  float* ws   = (float*)d_ws;
  float* h1   = ws + A_OFF;   // region A
  float* h2   = ws + B_OFF;   // region B
  float* e    = ws + E_OFF;   // region E
  float* gbuf = ws + G_OFF;   // region G
  float* xn   = h1;           // reuse A after conv3
  float* xcv  = h2;           // reuse B after conv3
  float* ybuf = h1;           // reuse A after gates
  float* xn2  = h2;           // reuse B after scan
  float* up   = gbuf;         // reuse G after scan

  k_conv1<<<dim3((P1 + 255) / 256, Bb), 256, 0, stream>>>(
      x, w1a, w1b, bn1_g, bn1_b, bn1_m, bn1_v, h1);
  k_conv2<<<dim3((P2 + 127) / 128, Bb), 128, 0, stream>>>(
      h1, w2, bn2_g, bn2_b, bn2_m, bn2_v, h2);
  k_conv3_wmma<<<dim3((L3 + 15) / 16, Bb), 256, 0, stream>>>(
      h2, w3, bn3_g, bn3_b, bn3_m, bn3_v, e);
  k_layernorm<<<Bb * S, 128, 0, stream>>>(e, xn, ln1_w);
  k_dwconv_silu<<<((size_t)Bb * S * D + 255) / 256, 256, 0, stream>>>(
      xn, conv_w, conv_b, xcv);
  k_gates_wmma<<<1020, 256, 0, stream>>>(xn, xcv, wi, wf, wz, wo, rbias, gbuf);
  k_scan<<<(Bb * H) / 8, 256, 0, stream>>>(gbuf, rkern, ybuf);
  k_gnorm_residual<<<Bb * S, 128, 0, stream>>>(ybuf, gn_w, e);
  k_layernorm<<<Bb * S, 128, 0, stream>>>(e, xn2, ln2_w);
  k_ffn_up_wmma<<<1020, 256, 0, stream>>>(xn2, ffn_up, up);
  k_geglu<<<((size_t)Bb * S * FF + 255) / 256, 256, 0, stream>>>(up);
  k_ffn_down_wmma<<<1020, 256, 0, stream>>>(up, ffn_dn, e);
  k_head<<<Bb, 128, 0, stream>>>(e, lnp_w, fc_w, fc_b, (float*)d_out);
}
